// Transformer_38010460570051
// MI455X (gfx1250) — compile-verified, matrix operand order checked by reading, not run on hardware
//
#include <hip/hip_runtime.h>
#include <hip/hip_bf16.h>

// ---------------------------------------------------------------------------
// Problem constants (B=2, H=W=64, C3=256, D = 256*9 = 2304, L = 4096)
// ---------------------------------------------------------------------------
#define BATCH 2
#define HH 64
#define WW 64
#define LPOS 4096           // H*W
#define DDIM 2304           // 256*9
#define CH3 256
#define KSTAGES 72          // DDIM / 32

typedef __attribute__((ext_vector_type(16))) __bf16 v16bf;
typedef __attribute__((ext_vector_type(8)))  float  v8f;
typedef unsigned long long u64;

union FragBF {
    v16bf v;
    u64 q[4];
};

// ---------------------------------------------------------------------------
// Kernel 1: per-position L2 norm of the 3x3x256 patch (one wave32 / position)
// ---------------------------------------------------------------------------
__global__ __launch_bounds__(256)
void norm_kernel(const float* __restrict__ q, const float* __restrict__ k,
                 float* __restrict__ invq, float* __restrict__ invk) {
    int wid  = (blockIdx.x * blockDim.x + threadIdx.x) >> 5;   // 0 .. B*L-1
    int lane = threadIdx.x & 31;
    if (wid >= BATCH * LPOS) return;
    int b = wid >> 12;
    int m = wid & (LPOS - 1);
    int y = m >> 6, x = m & 63;
    for (int t = 0; t < 2; ++t) {
        const float* src = t ? k : q;
        float s = 0.f;
        for (int d = lane; d < DDIM; d += 32) {
            int c = d / 9, j = d % 9;
            int yy = y + j / 3 - 1, xx = x + j % 3 - 1;
            if ((unsigned)yy < (unsigned)HH && (unsigned)xx < (unsigned)WW) {
                float v = src[(((size_t)b * CH3 + c) << 12) + (yy << 6) + xx];
                s += v * v;
            }
        }
        for (int o = 16; o > 0; o >>= 1) s += __shfl_xor(s, o, 32);
        float inv = 1.0f / fmaxf(sqrtf(s), 1e-12f);
        if (lane == 0) (t ? invk : invq)[wid] = inv;
    }
}

// ---------------------------------------------------------------------------
// Kernel 2: write normalized unfolded patches as bf16, layout [b][pos][d]
// ---------------------------------------------------------------------------
__global__ __launch_bounds__(256)
void pack_kernel(const float* __restrict__ src, const float* __restrict__ inv,
                 __bf16* __restrict__ dst) {
    size_t t = (size_t)blockIdx.x * blockDim.x + threadIdx.x;
    const size_t total = (size_t)BATCH * LPOS * DDIM;
    if (t >= total) return;
    int d   = (int)(t % DDIM);
    size_t pm = t / DDIM;                   // b*L + m
    int m = (int)(pm & (LPOS - 1));
    int b = (int)(pm >> 12);
    int c = d / 9, j = d % 9;
    int y = (m >> 6) + j / 3 - 1;
    int x = (m & 63) + j % 3 - 1;
    float v = 0.f;
    if ((unsigned)y < (unsigned)HH && (unsigned)x < (unsigned)WW)
        v = src[(((size_t)b * CH3 + c) << 12) + (y << 6) + x] * inv[pm];
    dst[t] = (__bf16)v;
}

// ---------------------------------------------------------------------------
// Kernel 3: corr = Kn * Qn^T fused with max/argmax over the key axis.
//   Tile: 128 keys x 64 queries, 128 threads (4 waves), wave owns 32x64
//   (2 A-frags x 4 B-frags -> 8 v_wmma_f32_16x16x32_bf16 per K=32 stage).
//   Double-buffered LDS filled by GLOBAL_LOAD_ASYNC_TO_LDS_B64 (ASYNCcnt):
//   async fills of buf[cur^1] overlap WMMA on buf[cur]; one
//   s_wait_asynccnt 0 + barrier per stage. Key axis split in 2 halves
//   (blockIdx.z) for occupancy; partials merged by merge_kernel.
// ---------------------------------------------------------------------------
__global__ __launch_bounds__(128)
void corr_kernel(const __bf16* __restrict__ Qn, const __bf16* __restrict__ Kn,
                 float* __restrict__ pv, int* __restrict__ pi) {
    __shared__ __align__(16) __bf16 As[2][128][36];   // keys    x 32 (+4 pad)
    __shared__ __align__(16) __bf16 Bs[2][64][36];    // queries x 32 (+4 pad)
    __shared__ float redv[4][64];
    __shared__ int   redi[4][64];

    const int b     = blockIdx.y;
    const int mtile = blockIdx.x;                  // 64 query-column blocks
    const int lhalf = blockIdx.z;                  // key-axis half (0/1)
    const __bf16* Kb = Kn + (size_t)b * LPOS * DDIM;
    const __bf16* Qb = Qn + (size_t)b * LPOS * DDIM + (size_t)mtile * 64 * DDIM;

    const int tid  = threadIdx.x;
    const int wave = tid >> 5;
    const int lane = tid & 31;
    const int ln16 = lane & 15;
    const int hf   = lane >> 4;

    float runv[4];
    int   runi[4];
#pragma unroll
    for (int i = 0; i < 4; ++i) { runv[i] = -3e38f; runi[i] = 0; }

    for (int lt = 0; lt < 16; ++lt) {              // 16 tiles of 128 key rows
        const int ltbase = (lhalf * 16 + lt) * 128;
        const __bf16* Ka = Kb + (size_t)ltbase * DDIM;

        // async fill of one K=32 stage (A: 128x32, B: 64x32) into buf
        auto issue_async = [&](int buf, int kc) {
#pragma unroll
            for (int it = 0; it < 8; ++it) {
                int slot = tid + it * 128, row = slot >> 3, cg = slot & 7;
                unsigned lds = (unsigned)(size_t)&As[buf][row][cg * 4];
                const __bf16* g = Ka + (size_t)row * DDIM + kc + cg * 4;
                asm volatile("global_load_async_to_lds_b64 %0, %1, off"
                             :: "v"(lds), "v"(g) : "memory");
            }
#pragma unroll
            for (int it = 0; it < 4; ++it) {
                int slot = tid + it * 128, row = slot >> 3, cg = slot & 7;
                unsigned lds = (unsigned)(size_t)&Bs[buf][row][cg * 4];
                const __bf16* g = Qb + (size_t)row * DDIM + kc + cg * 4;
                asm volatile("global_load_async_to_lds_b64 %0, %1, off"
                             :: "v"(lds), "v"(g) : "memory");
            }
        };

        v8f cc[2][4];
#pragma unroll
        for (int mf = 0; mf < 2; ++mf)
#pragma unroll
            for (int nf = 0; nf < 4; ++nf)
#pragma unroll
                for (int i = 0; i < 8; ++i) cc[mf][nf][i] = 0.f;

        issue_async(0, 0);                         // prologue: stage 0

        for (int s = 0; s < KSTAGES; ++s) {
            const int cur = s & 1;
            asm volatile("s_wait_asynccnt 0x0" ::: "memory");
            __syncthreads();                       // buf[cur] complete+visible
            if (s + 1 < KSTAGES)                   // fill other buffer during
                issue_async(cur ^ 1, (s + 1) * 32);//   this stage's WMMAs

            // preload all fragments into distinct registers, then 8 WMMAs
            FragBF fa[2], fb[4];
#pragma unroll
            for (int mf = 0; mf < 2; ++mf) {
                const u64* r = (const u64*)&As[cur][wave * 32 + mf * 16 + ln16][0];
                fa[mf].q[0] = r[2 * hf];
                fa[mf].q[1] = r[2 * hf + 1];
                fa[mf].q[2] = r[4 + 2 * hf];
                fa[mf].q[3] = r[5 + 2 * hf];
            }
#pragma unroll
            for (int nf = 0; nf < 4; ++nf) {
                const u64* r = (const u64*)&Bs[cur][nf * 16 + ln16][0];
                fb[nf].q[0] = r[2 * hf];
                fb[nf].q[1] = r[2 * hf + 1];
                fb[nf].q[2] = r[4 + 2 * hf];
                fb[nf].q[3] = r[5 + 2 * hf];
            }
#pragma unroll
            for (int nf = 0; nf < 4; ++nf) {
                cc[0][nf] = __builtin_amdgcn_wmma_f32_16x16x32_bf16(
                    false, fa[0].v, false, fb[nf].v, (short)0, cc[0][nf], false, false);
                cc[1][nf] = __builtin_amdgcn_wmma_f32_16x16x32_bf16(
                    false, fa[1].v, false, fb[nf].v, (short)0, cc[1][nf], false, false);
            }
        }

        // ---- epilogue: running max/argmax over this tile's 32 rows/wave ----
#pragma unroll
        for (int nf = 0; nf < 4; ++nf) {
            float bv = -3e38f;
            int   bi = 0;
#pragma unroll
            for (int mf = 0; mf < 2; ++mf)
#pragma unroll
                for (int r = 0; r < 8; ++r) {
                    float v = cc[mf][nf][r];
                    int l = ltbase + wave * 32 + mf * 16 + 8 * hf + r;
                    if (v > bv) { bv = v; bi = l; }
                }
            float ov = __shfl_xor(bv, 16, 32);     // partner holds other 8 rows
            int   oi = __shfl_xor(bi, 16, 32);
            if (ov > bv || (ov == bv && oi < bi)) { bv = ov; bi = oi; }
            if (bv > runv[nf]) { runv[nf] = bv; runi[nf] = bi; }
        }
    }

    // cross-wave reduction via LDS
    if (lane < 16) {
#pragma unroll
        for (int nf = 0; nf < 4; ++nf) {
            redv[wave][nf * 16 + ln16] = runv[nf];
            redi[wave][nf * 16 + ln16] = runi[nf];
        }
    }
    __syncthreads();
    if (tid < 64) {
        float bv = redv[0][tid];
        int   bi = redi[0][tid];
#pragma unroll
        for (int w = 1; w < 4; ++w) {
            float v = redv[w][tid];
            int   i = redi[w][tid];
            if (v > bv || (v == bv && i < bi)) { bv = v; bi = i; }
        }
        size_t o = (size_t)lhalf * BATCH * LPOS + (size_t)b * LPOS + mtile * 64 + tid;
        pv[o] = bv;
        pi[o] = bi;
    }
}

// ---------------------------------------------------------------------------
// Kernel 3b: merge the two key-axis halves -> S (d_out) and final indices.
//   Half 0 always holds the smaller key index, so ties keep half 0.
// ---------------------------------------------------------------------------
__global__ __launch_bounds__(256)
void merge_kernel(const float* __restrict__ pv, const int* __restrict__ pi,
                  float* __restrict__ S, int* __restrict__ idx) {
    int t = blockIdx.x * blockDim.x + threadIdx.x;
    if (t >= BATCH * LPOS) return;
    float v0 = pv[t], v1 = pv[BATCH * LPOS + t];
    int   i0 = pi[t], i1 = pi[BATCH * LPOS + t];
    bool take1 = v1 > v0;
    S[t]   = take1 ? v1 : v0;
    idx[t] = take1 ? i1 : i0;
}

// ---------------------------------------------------------------------------
// Kernel 4: fused gather + fold for one pyramid level (scale s = 1,2,4,8).
//   T[b,c,y,x] = (1/9) * sum over covering patches m=(qy+dy, qx+dx):
//       V[b,c, y + (l_y - m_y)*s, x + (l_x - m_x)*s]   (OOB -> 0)
// ---------------------------------------------------------------------------
__global__ __launch_bounds__(256)
void fold_kernel(const float* __restrict__ V, const int* __restrict__ idx,
                 float* __restrict__ out, int C, int s, int HWs) {
    size_t t = (size_t)blockIdx.x * blockDim.x + threadIdx.x;
    size_t total = (size_t)BATCH * C * HWs * HWs;
    if (t >= total) return;
    int x = (int)(t % HWs);
    size_t r = t / HWs;
    int y = (int)(r % HWs); r /= HWs;
    int c = (int)(r % C);
    int b = (int)(r / C);
    int qy = y / s, qx = x / s;
    const int* idxb = idx + (size_t)b * LPOS;
    const float* Vb = V + ((size_t)b * C + c) * HWs * HWs;
    float acc = 0.f;
#pragma unroll
    for (int dy = -1; dy <= 1; ++dy) {
        int my = qy + dy;
        if ((unsigned)my >= 64u) continue;
#pragma unroll
        for (int dx = -1; dx <= 1; ++dx) {
            int mx = qx + dx;
            if ((unsigned)mx >= 64u) continue;
            int l  = idxb[my * 64 + mx];
            int ly = l >> 6, lx = l & 63;
            int sy = y + (ly - my) * s;
            int sx = x + (lx - mx) * s;
            if ((unsigned)sy < (unsigned)HWs && (unsigned)sx < (unsigned)HWs)
                acc += Vb[(size_t)sy * HWs + sx];
        }
    }
    out[t] = acc * (1.0f / 9.0f);
}

// ---------------------------------------------------------------------------
// Launcher
// ---------------------------------------------------------------------------
extern "C" void kernel_launch(void* const* d_in, const int* in_sizes, int n_in,
                              void* d_out, int out_size, void* d_ws, size_t ws_size,
                              hipStream_t stream) {
    const float* q  = (const float*)d_in[0];   // (2,256,64,64)
    const float* k  = (const float*)d_in[1];   // (2,256,64,64)
    const float* v0 = (const float*)d_in[2];   // (2,32,512,512)
    const float* v1 = (const float*)d_in[3];   // (2,64,256,256)
    const float* v2 = (const float*)d_in[4];   // (2,128,128,128)
    const float* v3 = (const float*)d_in[5];   // (2,256,64,64)
    float* out = (float*)d_out;

    // workspace carve-up
    const size_t qk_elems = (size_t)BATCH * LPOS * DDIM;   // 18,874,368
    __bf16* Qn = reinterpret_cast<__bf16*>(d_ws);
    __bf16* Kn = Qn + qk_elems;
    float* invq = reinterpret_cast<float*>(Kn + qk_elems);
    float* invk = invq + (size_t)BATCH * LPOS;
    int*   idxb = reinterpret_cast<int*>(invk + (size_t)BATCH * LPOS);
    float* pv   = reinterpret_cast<float*>(idxb + (size_t)BATCH * LPOS);
    int*   pi   = reinterpret_cast<int*>(pv + (size_t)2 * BATCH * LPOS);

    // 1) per-position patch norms (one wave32 per position, 8 waves/block)
    norm_kernel<<<(BATCH * LPOS) / 8, 256, 0, stream>>>(q, k, invq, invk);

    // 2) pack normalized unfolded patches to bf16
    {
        int grid = (int)((qk_elems + 255) / 256);
        pack_kernel<<<grid, 256, 0, stream>>>(q, invq, Qn);
        pack_kernel<<<grid, 256, 0, stream>>>(k, invk, Kn);
    }

    // 3) WMMA correlation GEMM + fused max/argmax over split key axis
    {
        dim3 grid(LPOS / 64, BATCH, 2);
        corr_kernel<<<grid, 128, 0, stream>>>(Qn, Kn, pv, pi);
        merge_kernel<<<(BATCH * LPOS) / 256, 256, 0, stream>>>(pv, pi, out, idxb);
    }

    // 4) gather+fold pyramid levels (output offsets in return order)
    float* T3 = out + 8192;
    float* T2 = T3 + (size_t)BATCH * 256 * 64 * 64;     // 2,097,152
    float* T1 = T2 + (size_t)BATCH * 128 * 128 * 128;   // 4,194,304
    float* T0 = T1 + (size_t)BATCH * 64 * 256 * 256;    // 8,388,608

    fold_kernel<<<(int)(((size_t)BATCH * 256 * 64 * 64 + 255) / 256), 256, 0, stream>>>(
        v3, idxb, T3, 256, 1, 64);
    fold_kernel<<<(int)(((size_t)BATCH * 128 * 128 * 128 + 255) / 256), 256, 0, stream>>>(
        v2, idxb, T2, 128, 2, 128);
    fold_kernel<<<(int)(((size_t)BATCH * 64 * 256 * 256 + 255) / 256), 256, 0, stream>>>(
        v1, idxb, T1, 64, 4, 256);
    fold_kernel<<<(int)(((size_t)BATCH * 32 * 512 * 512 + 255) / 256), 256, 0, stream>>>(
        v0, idxb, T0, 32, 8, 512);
}
